// Net_69715909148679
// MI455X (gfx1250) — matrix operand order repaired and checked by reading, not compiled, not run on hardware
//
#include <hip/hip_runtime.h>
#include <math.h>

// ---------------------------------------------------------------------------
// GIN network for MI455X (gfx1250). All GEMMs use v_wmma_f32_16x16x32_f16
// (f16 inputs, f32 accumulate). BN folded into preceding Linear. Edge
// aggregation via L2-resident f32 atomics (h1/agg2 fit in 192MB L2).
// ---------------------------------------------------------------------------

typedef __attribute__((ext_vector_type(16))) _Float16 v16h;
typedef __attribute__((ext_vector_type(8)))  float    v8f;

#define BN_EPS 1e-5f

// ---- WMMA fragment loaders (ISA 7.12.2 layouts, wave32) --------------------
// A matrix 16x32 f16: lane (g=lane/16, r=lane%16) holds row r;
//   elements 0..7  = K[k0+8g .. k0+8g+7]      (contiguous 16B)
//   elements 8..15 = K[k0+16+8g .. k0+23+8g]  (contiguous 16B)
__device__ inline v16h ldA16(const _Float16* rowbase, int ld, int k0, int lane) {
  int r = lane & 15, g = lane >> 4;
  const _Float16* p = rowbase + (size_t)r * ld + k0 + 8 * g;
  union { v16h h; float4 f[2]; } u;
  u.f[0] = *(const float4*)(p);
  u.f[1] = *(const float4*)(p + 16);
  return u.h;
}

// B matrix 32x16 f16 pre-packed so each lane reads 16 contiguous halves.
__device__ inline v16h ldB16(const _Float16* p) {
  union { v16h h; float4 f[2]; } u;
  u.f[0] = *(const float4*)(p);
  u.f[1] = *(const float4*)(p + 8);
  return u.h;
}

__device__ inline v8f wmma_f16(v16h a, v16h b, v8f c) {
  return __builtin_amdgcn_wmma_f32_16x16x32_f16(false, a, false, b, (short)0, c,
                                                false, false);
}

// ---- Fold BN into linear coefficients --------------------------------------
__global__ void k_coeffs(const float* w1a, const float* b1a, const float* g1,
                         const float* be1, const float* m1, const float* v1,
                         const float* b2a, const float* g2, const float* be2,
                         const float* m2, const float* v2,
                         float* A1, float* B1, float* S2, float* bias2) {
  int c = threadIdx.x;
  if (c < 256) {
    float s1 = g1[c] * rsqrtf(v1[c] + BN_EPS);
    A1[c] = w1a[c] * s1;
    B1[c] = (b1a[c] - m1[c]) * s1 + be1[c];
  }
  if (c < 64) {
    float s2 = g2[c] * rsqrtf(v2[c] + BN_EPS);
    S2[c] = s2;
    bias2[c] = (b2a[c] - m2[c]) * s2 + be2[c];
  }
}

// ---- Pre-pack a row-major f32 [Ks x Ns] weight into B-fragment order -------
// dst[frag*512 + lane*16 + j] = w[kstep*32 + (j<8 ? 8*(lane/16)+j : 16+8*(lane/16)+j-8)][ntile*16 + lane%16]
__global__ void k_prepack(const float* src, int Ks, int Ns, int ksteps,
                          int ntiles, const float* colscale, _Float16* dst) {
  int total = ksteps * ntiles * 512;
  for (int t = blockIdx.x * blockDim.x + threadIdx.x; t < total;
       t += gridDim.x * blockDim.x) {
    int frag = t >> 9;
    int rem = t & 511;
    int lane = rem >> 4;
    int j = rem & 15;
    int g = lane >> 4, n = lane & 15;
    int kstep = frag / ntiles, ntile = frag % ntiles;
    int K = kstep * 32 + (j < 8 ? 8 * g + j : 16 + 8 * g + (j - 8));
    int col = ntile * 16 + n;
    float v = 0.f;
    if (K < Ks && col < Ns) {
      v = src[(size_t)K * Ns + col];
      if (colscale) v *= colscale[col];
    }
    dst[t] = (_Float16)v;
  }
}

__global__ void k_zero(float* p, long long n) {
  long long t = blockIdx.x * (long long)blockDim.x + threadIdx.x;
  long long s = (long long)gridDim.x * blockDim.x;
  for (; t < n; t += s) p[t] = 0.f;
}

// ---- Layer-1 edge scatter: scalar feature ----------------------------------
__global__ void k_edge1(const int* srcI, const int* dstI, const float* x,
                        float* agg1, int E) {
  int e = blockIdx.x * blockDim.x + threadIdx.x;
  if (e < E) atomicAdd(&agg1[dstI[e]], x[srcI[e]]);
}

// ---- Fused GIN1: s=x+agg1 -> BN/ReLU act (16x256) -> WMMA @ w1b -> h1 ------
__global__ void __launch_bounds__(256)
k_gin1(const float* x, const float* agg1, const float* A1, const float* B1,
       const _Float16* packB, const float* b1b, float* h1, int Nn) {
  __shared__ float sS[16];
  __shared__ __align__(16) _Float16 sAct[16][256];  // 8KB
  int tid = threadIdx.x;
  int base = blockIdx.x * 16;
  if (tid < 16) {
    int row = base + tid;
    sS[tid] = (row < Nn) ? (x[row] + agg1[row]) : 0.f;
  }
  __syncthreads();
  {
    float a = A1[tid], b = B1[tid];
#pragma unroll
    for (int i = 0; i < 16; ++i)
      sAct[i][tid] = (_Float16)fmaxf(sS[i] * a + b, 0.f);
  }
  __syncthreads();
  int w = tid >> 5, lane = tid & 31;  // 8 waves, one 16-col tile each
  v8f acc = {};
#pragma unroll
  for (int ks = 0; ks < 8; ++ks) {
    v16h a = ldA16(&sAct[0][0], 256, ks * 32, lane);
    v16h b = ldB16(packB + (((ks << 3) + w) << 9) + lane * 16);
    acc = wmma_f16(a, b, acc);
  }
  int n = lane & 15, g = lane >> 4;
  int col = w * 16 + n;
  float bias = b1b[col];
#pragma unroll
  for (int j = 0; j < 8; ++j) {
    int row = base + j + 8 * g;
    if (row < Nn) h1[(size_t)row * 128 + col] = acc[j] + bias;
  }
}

// ---- Layer-2 edge scatter: one wave per edge, 4 coalesced atomics/lane -----
__global__ void k_edge2(const int* srcI, const int* dstI, const float* h1,
                        float* agg2, int E) {
  long long t = blockIdx.x * (long long)blockDim.x + threadIdx.x;
  int e = (int)(t >> 5);
  int lane = (int)(t & 31);
  if (e < E) {
    const float* hp = h1 + (size_t)srcI[e] * 128;
    float* ap = agg2 + (size_t)dstI[e] * 128;
#pragma unroll
    for (int k = 0; k < 4; ++k) {
      int c = lane + 32 * k;
      atomicAdd(&ap[c], hp[c]);
    }
  }
}

// ---- Fused tail: (h1+agg2) -> mlp2a(BN,ReLU) -> mlp2b -> head1(ReLU) ->
//      head2 (K padded 16->32) -> log_softmax. Four chained WMMA GEMMs. ------
__global__ void __launch_bounds__(128)
k_tail(const float* h1, const float* agg2, const _Float16* pW2a,
       const float* bias2, const _Float16* pW2b, const float* b2b,
       const _Float16* pWl1, const float* bl1, const _Float16* pWl2,
       const float* bl2, float* out, int Nn) {
  __shared__ __align__(16) _Float16 sA[16][128];
  __shared__ __align__(16) _Float16 sT1[16][64];
  __shared__ __align__(16) _Float16 sX2[16][64];
  __shared__ __align__(16) _Float16 sT3[16][32];
  __shared__ float sLg[16][16];
  int tid = threadIdx.x;
  int base = blockIdx.x * 16;
#pragma unroll
  for (int i = 0; i < 16; ++i) {
    int row = base + i;
    float v = 0.f;
    if (row < Nn) {
      size_t idx = (size_t)row * 128 + tid;
      v = h1[idx] + agg2[idx];
    }
    sA[i][tid] = (_Float16)v;
  }
  __syncthreads();
  int w = tid >> 5, lane = tid & 31;  // 4 waves
  int n = lane & 15, g = lane >> 4;
  {  // mlp2a: [16x128]@[128x64], fused BN scale already in pW2a
    v8f acc = {};
#pragma unroll
    for (int ks = 0; ks < 4; ++ks) {
      v16h a = ldA16(&sA[0][0], 128, ks * 32, lane);
      v16h b = ldB16(pW2a + (((ks << 2) + w) << 9) + lane * 16);
      acc = wmma_f16(a, b, acc);
    }
    int col = w * 16 + n;
    float bs = bias2[col];
#pragma unroll
    for (int j = 0; j < 8; ++j)
      sT1[j + 8 * g][col] = (_Float16)fmaxf(acc[j] + bs, 0.f);
  }
  __syncthreads();
  {  // mlp2b: [16x64]@[64x64]
    v8f acc = {};
#pragma unroll
    for (int ks = 0; ks < 2; ++ks) {
      v16h a = ldA16(&sT1[0][0], 64, ks * 32, lane);
      v16h b = ldB16(pW2b + (((ks << 2) + w) << 9) + lane * 16);
      acc = wmma_f16(a, b, acc);
    }
    int col = w * 16 + n;
    float bs = b2b[col];
#pragma unroll
    for (int j = 0; j < 8; ++j)
      sX2[j + 8 * g][col] = (_Float16)(acc[j] + bs);
  }
  __syncthreads();
  if (w == 0) {  // head1: [16x64]@[64x16] + ReLU, pad K-columns 16..31 with 0
    v8f acc = {};
#pragma unroll
    for (int ks = 0; ks < 2; ++ks) {
      v16h a = ldA16(&sX2[0][0], 64, ks * 32, lane);
      v16h b = ldB16(pWl1 + (ks << 9) + lane * 16);
      acc = wmma_f16(a, b, acc);
    }
    float bs = bl1[n];
#pragma unroll
    for (int j = 0; j < 8; ++j) {
      sT3[j + 8 * g][n] = (_Float16)fmaxf(acc[j] + bs, 0.f);
      sT3[j + 8 * g][n + 16] = (_Float16)0.f;
    }
  }
  __syncthreads();
  if (w == 0) {  // head2: [16x32(pad)]@[32x16(pad)] -> logits in cols 0..5
    v8f acc = {};
    v16h a = ldA16(&sT3[0][0], 32, 0, lane);
    v16h b = ldB16(pWl2 + lane * 16);
    acc = wmma_f16(a, b, acc);
    float bs = (n < 6) ? bl2[n] : 0.f;
#pragma unroll
    for (int j = 0; j < 8; ++j) sLg[j + 8 * g][n] = acc[j] + bs;
  }
  __syncthreads();
  if (tid < 16) {
    int row = base + tid;
    if (row < Nn) {
      float m = sLg[tid][0];
#pragma unroll
      for (int c = 1; c < 6; ++c) m = fmaxf(m, sLg[tid][c]);
      float sum = 0.f;
#pragma unroll
      for (int c = 0; c < 6; ++c) sum += expf(sLg[tid][c] - m);
      float ls = logf(sum) + m;
#pragma unroll
      for (int c = 0; c < 6; ++c) out[(size_t)row * 6 + c] = sLg[tid][c] - ls;
    }
  }
}

// ---------------------------------------------------------------------------
extern "C" void kernel_launch(void* const* d_in, const int* in_sizes, int n_in,
                              void* d_out, int out_size, void* d_ws,
                              size_t ws_size, hipStream_t stream) {
  const float* x = (const float*)d_in[0];
  const int* ei = (const int*)d_in[1];
  int Nn = in_sizes[0];
  int E = in_sizes[1] / 2;
  const int* srcI = ei;
  const int* dstI = ei + E;
  const float* w1a = (const float*)d_in[2];
  const float* b1a = (const float*)d_in[3];
  const float* g1  = (const float*)d_in[4];
  const float* be1 = (const float*)d_in[5];
  const float* m1  = (const float*)d_in[6];
  const float* v1  = (const float*)d_in[7];
  const float* w1b = (const float*)d_in[8];
  const float* b1b = (const float*)d_in[9];
  const float* w2a = (const float*)d_in[10];
  const float* b2a = (const float*)d_in[11];
  const float* g2  = (const float*)d_in[12];
  const float* be2 = (const float*)d_in[13];
  const float* m2  = (const float*)d_in[14];
  const float* v2  = (const float*)d_in[15];
  const float* w2b = (const float*)d_in[16];
  const float* b2b = (const float*)d_in[17];
  const float* wl1 = (const float*)d_in[18];
  const float* bl1 = (const float*)d_in[19];
  const float* wl2 = (const float*)d_in[20];
  const float* bl2 = (const float*)d_in[21];

  char* ws = (char*)d_ws;
  size_t off = 0;
  auto take = [&](size_t bytes) -> void* {
    void* p = ws + off;
    off = (off + bytes + 255) & ~(size_t)255;
    return p;
  };
  float* agg1  = (float*)take((size_t)Nn * 4);
  float* h1    = (float*)take((size_t)Nn * 128 * 4);
  float* agg2  = (float*)take((size_t)Nn * 128 * 4);
  float* A1    = (float*)take(256 * 4);
  float* B1    = (float*)take(256 * 4);
  float* S2    = (float*)take(64 * 4);
  float* bias2 = (float*)take(64 * 4);
  _Float16* pW1b = (_Float16*)take(32768 * 2);  // 8 ksteps x 8 ntiles
  _Float16* pW2a = (_Float16*)take(8192 * 2);   // 4 x 4
  _Float16* pW2b = (_Float16*)take(4096 * 2);   // 2 x 4
  _Float16* pWl1 = (_Float16*)take(1024 * 2);   // 2 x 1
  _Float16* pWl2 = (_Float16*)take(512 * 2);    // 1 x 1 (padded)

  k_coeffs<<<1, 256, 0, stream>>>(w1a, b1a, g1, be1, m1, v1, b2a, g2, be2, m2,
                                  v2, A1, B1, S2, bias2);
  k_prepack<<<64, 256, 0, stream>>>(w1b, 256, 128, 8, 8, nullptr, pW1b);
  k_prepack<<<16, 256, 0, stream>>>(w2a, 128, 64, 4, 4, S2, pW2a);
  k_prepack<<<8, 256, 0, stream>>>(w2b, 64, 64, 2, 4, nullptr, pW2b);
  k_prepack<<<2, 256, 0, stream>>>(wl1, 64, 16, 2, 1, nullptr, pWl1);
  k_prepack<<<1, 256, 0, stream>>>(wl2, 16, 6, 1, 1, nullptr, pWl2);

  k_zero<<<512, 256, 0, stream>>>(agg1, (long long)Nn);
  k_edge1<<<(E + 255) / 256, 256, 0, stream>>>(srcI, dstI, x, agg1, E);

  int mtiles = (Nn + 15) / 16;
  k_gin1<<<mtiles, 256, 0, stream>>>(x, agg1, A1, B1, pW1b, b1b, h1, Nn);

  k_zero<<<4096, 256, 0, stream>>>(agg2, (long long)Nn * 128);
  long long t2 = (long long)E * 32;
  k_edge2<<<(unsigned)((t2 + 255) / 256), 256, 0, stream>>>(srcI, dstI, h1,
                                                            agg2, E);

  k_tail<<<mtiles, 128, 0, stream>>>(h1, agg2, pW2a, bias2, pW2b, b2b, pWl1,
                                     bl1, pWl2, bl2, (float*)d_out, Nn);
}
